// ResonanceLayer_43181601193992
// MI455X (gfx1250) — matrix operand order repaired and checked by reading, not compiled, not run on hardware
//
#include <hip/hip_runtime.h>
#include <math.h>
#include <stdint.h>

// MI455X / gfx1250, wave32. One workgroup (256 thr = 8 waves) per batch b.
// LDS layout (dynamic):
//   [0      .. 65536)  A  : 256x128 f16  (fc1 input)   -- later reused as Fre: 256x64 f32
//   [65536  .. 131072) H1 : 256x128 f16  (fc1 output / fc2 input)
//   [131072 .. +10176) small: te_w, te_b, haar, pid/dist/ang, counts/pos

typedef __attribute__((ext_vector_type(16))) _Float16 v16h;
typedef __attribute__((ext_vector_type(8)))  float    v8f;

union AFrag { v16h v; uint32_t u[8]; };
union BFrag { v16h v; _Float16 h[16]; };

#define LDS_A_OFF   0
#define LDS_H1_OFF  65536
#define LDS_SM_OFF  131072

// small-region word offsets (floats)
#define SM_TEW   0      // 512 f32 : te_w (4x128)
#define SM_TEB   512    // 128 f32 : te_b
#define SM_HAAR  640    // 1040 f32: 260 rows x 4 haar coeffs (row 0 = ego)
#define SM_PID   1680   // 256 i32 : partition id per (n,t)  (-1 invalid)
#define SM_DIST  1936   // 256 f32
#define SM_ANG   2192   // 256 f32
#define SM_CNT   2448   // 32 f32  : count + 1e-4 per (t,p)
#define SM_POSD  2480   // 32 f32
#define SM_POSA  2512   // 32 f32
#define SM_WORDS 2544
#define LDS_BYTES (131072 + SM_WORDS * 4)

// A-matrix (M x K = 16x32, f16) fragment from LDS (row-major, ld halfs, k0 mult of 32).
// ISA 16-bit A layout: lanes 0-15: VGPR0..3 = K{0..7}, VGPR4..7 = K{16..23};
//                      lanes 16-31: K{8..15}, K{24..31}.
__device__ __forceinline__ void load_a_frag(const uint32_t* L, int m0, int k0, int ld,
                                            int lane, AFrag& af) {
  const int m = lane & 15, half = lane >> 4;
#pragma unroll
  for (int v = 0; v < 8; ++v) {
    const int kb = ((v < 4) ? (2 * v) : (16 + 2 * (v - 4))) + 8 * half;
    af.u[v] = L[((m0 + m) * ld + k0 + kb) >> 1];   // two consecutive f16
  }
}

// B-matrix (K x N = 32x16) fragment from GLOBAL f32 weights (row-major K x ldn), cvt to f16.
// Dense-B layout (matches A): lanes 0-15: VGPR0..3 = K{0..7}, VGPR4..7 = K{16..23};
// lanes 16-31: K{8..15}, K{24..31}.
__device__ __forceinline__ void load_b_frag_g(const float* __restrict__ W, int k0, int n0,
                                              int ldn, int lane, BFrag& bf) {
  const int n = lane & 15, half = lane >> 4;
#pragma unroll
  for (int v = 0; v < 8; ++v) {
    const int k = k0 + ((v < 4) ? (2 * v) : (16 + 2 * (v - 4))) + 8 * half;
    bf.h[2 * v]     = (_Float16)W[k * ldn + n0 + n];
    bf.h[2 * v + 1] = (_Float16)W[(k + 1) * ldn + n0 + n];
  }
}

__global__ __launch_bounds__(256)
void resonance_kernel(const float* __restrict__ x_ego, const float* __restrict__ x_nei,
                      const float* __restrict__ te_w, const float* __restrict__ te_b,
                      const float* __restrict__ fc1_w, const float* __restrict__ fc1_b,
                      const float* __restrict__ fc2_w, const float* __restrict__ fc2_b,
                      const float* __restrict__ ce_w,  const float* __restrict__ ce_b,
                      float* __restrict__ out) {
  extern __shared__ char smem[];
  _Float16* Alds   = (_Float16*)(smem + LDS_A_OFF);
  uint32_t* Alds32 = (uint32_t*)(smem + LDS_A_OFF);
  _Float16* H1     = (_Float16*)(smem + LDS_H1_OFF);
  uint32_t* H1u32  = (uint32_t*)(smem + LDS_H1_OFF);
  float*    Fre    = (float*)(smem + LDS_A_OFF);     // aliases A after fc1
  float*    sm     = (float*)(smem + LDS_SM_OFF);
  float* s_tew = sm + SM_TEW;
  float* s_teb = sm + SM_TEB;
  float* s_haar = sm + SM_HAAR;
  int*   s_pid = (int*)(sm + SM_PID);
  float* s_dist = sm + SM_DIST;
  float* s_ang = sm + SM_ANG;
  float* s_cnt = sm + SM_CNT;
  float* s_posd = sm + SM_POSD;
  float* s_posa = sm + SM_POSA;

  const int tid  = threadIdx.x;
  const int b    = blockIdx.x;
  const int lane = tid & 31;
  const int wave = tid >> 5;
  const float INV_SQRT2 = 0.70710678118654752f;
  const float TWO_PI = 6.283185307179586f;

  // ---- load te weights ----
  for (int i = tid; i < 512; i += 256) s_tew[i] = te_w[i];
  if (tid < 128) s_teb[tid] = te_b[tid];

  const float* egoB = x_ego + (size_t)b * 16;          // (T=8, 2)
  const float* neiB = x_nei + (size_t)b * 1024;        // (N=64, T=8, 2)

  // ---- haar coefficients: pack row 0 = ego, rows 1..64 = neighbors ----
  for (int idx = tid; idx < 260; idx += 256) {
    const int n = idx >> 2, j = idx & 3;
    const float* src = (n == 0) ? egoB : (neiB + (n - 1) * 16);
    const float lx = src[14], ly = src[15];            // t=7 (subtracted "pure" ref)
    const float ax = src[4 * j + 0] - lx, ay = src[4 * j + 1] - ly;   // t=2j
    const float bx = src[4 * j + 2] - lx, by = src[4 * j + 3] - ly;   // t=2j+1
    s_haar[idx * 4 + 0] = (ax + bx) * INV_SQRT2;
    s_haar[idx * 4 + 1] = (ay + by) * INV_SQRT2;
    s_haar[idx * 4 + 2] = (ax - bx) * INV_SQRT2;
    s_haar[idx * 4 + 3] = (ay - by) * INV_SQRT2;
  }

  // ---- per-(n,t) distance / angle / partition id (t' -> original t=2t') ----
  {
    const int n = tid >> 2, t = tid & 3;
    const float* src = neiB + n * 16;
    const float dx = src[4 * t + 0], dy = src[4 * t + 1];
    const float dist = sqrtf(dx * dx + dy * dy);
    float ang = atan2f(dx, dy);                        // arctan2(x, y) per reference
    if (ang < 0.f) ang += TWO_PI;
    int pid = (int)(ang * 1.2732395447351628f);        // / (2pi/8)
    if (pid > 7) pid = 7;
    const bool valid = ((dx + dy) != 0.f) && ((src[14] + src[15]) != 0.f);
    if (!valid) pid = -1;
    s_pid[tid] = pid; s_dist[tid] = dist; s_ang[tid] = ang;
  }
  __syncthreads();

  // ---- A = f16( relu(te(ego)) * relu(te(nei)) ) : 256 rows (n*4+t) x 128 ----
  for (int i = tid; i < 32768; i += 256) {
    const int r = i >> 7, k = i & 127;
    const int n = r >> 2, j = r & 3;
    const float* he = s_haar + j * 4;                      // ego pack row 0
    const float* hn = s_haar + ((n + 1) * 4 + j) * 4;
    float fe = s_teb[k], fn = fe;
#pragma unroll
    for (int q = 0; q < 4; ++q) {
      const float w = s_tew[q * 128 + k];
      fe += he[q] * w;
      fn += hn[q] * w;
    }
    fe = fmaxf(fe, 0.f); fn = fmaxf(fn, 0.f);
    Alds[i] = (_Float16)(fe * fn);
  }
  __syncthreads();

  // ---- fc1: (256x128) @ (128x128) + b, ReLU -> H1 (f16). Wave w owns N-tile w. ----
  {
    const int n0 = wave * 16;
    const int half = lane >> 4;
    BFrag bfr[4];
#pragma unroll
    for (int kc = 0; kc < 4; ++kc) load_b_frag_g(fc1_w, kc * 32, n0, 128, lane, bfr[kc]);
    const int col = n0 + (lane & 15);
    const float bias = fc1_b[col];
    for (int mt = 0; mt < 16; ++mt) {
      const int m0 = mt * 16;
      v8f acc = {0.f, 0.f, 0.f, 0.f, 0.f, 0.f, 0.f, 0.f};
#pragma unroll
      for (int kc = 0; kc < 4; ++kc) {
        AFrag af;
        load_a_frag(Alds32, m0, kc * 32, 128, lane, af);
        acc = __builtin_amdgcn_wmma_f32_16x16x32_f16(false, af.v, false, bfr[kc].v,
                                                     (short)0, acc, false, false);
      }
#pragma unroll
      for (int j = 0; j < 8; ++j) {
        const int row = m0 + j + 8 * half;
        H1[row * 128 + col] = (_Float16)fmaxf(acc[j] + bias, 0.f);
      }
    }
  }
  __syncthreads();

  // ---- fc2: (256x128) @ (128x64) + b, ReLU -> Fre (f32, overwrites A region) ----
  {
    const int n0 = (wave & 3) * 16;
    const int mbase = (wave >> 2) * 8;
    const int half = lane >> 4;
    BFrag bfr[4];
#pragma unroll
    for (int kc = 0; kc < 4; ++kc) load_b_frag_g(fc2_w, kc * 32, n0, 64, lane, bfr[kc]);
    const int col = n0 + (lane & 15);
    const float bias = fc2_b[col];
    for (int mi = 0; mi < 8; ++mi) {
      const int m0 = (mbase + mi) * 16;
      v8f acc = {0.f, 0.f, 0.f, 0.f, 0.f, 0.f, 0.f, 0.f};
#pragma unroll
      for (int kc = 0; kc < 4; ++kc) {
        AFrag af;
        load_a_frag(H1u32, m0, kc * 32, 128, lane, af);
        acc = __builtin_amdgcn_wmma_f32_16x16x32_f16(false, af.v, false, bfr[kc].v,
                                                     (short)0, acc, false, false);
      }
#pragma unroll
      for (int j = 0; j < 8; ++j) {
        const int row = m0 + j + 8 * half;
        Fre[row * 64 + col] = fmaxf(acc[j] + bias, 0.f);
      }
    }
  }
  __syncthreads();

  // ---- write f_re (second output): out[4194304 + b*16384 ...], flat copy ----
  {
    const float4* src4 = (const float4*)Fre;
    float4* dst4 = (float4*)(out + 4194304 + (size_t)b * 16384);
    for (int i = tid; i < 4096; i += 256) dst4[i] = src4[i];
  }

  // ---- per-(t,p) counts / mean distance / mean angle ----
  if (tid < 32) {
    const int t = tid >> 3, p = tid & 7;
    float cnt = 0.f, sd = 0.f, sa = 0.f;
    for (int n = 0; n < 64; ++n) {
      if (s_pid[n * 4 + t] == p) {
        cnt += 1.f; sd += s_dist[n * 4 + t]; sa += s_ang[n * 4 + t];
      }
    }
    const float nn = cnt + 1e-4f;
    s_cnt[tid] = nn;
    s_posd[tid] = sd / nn;
    s_posa[tid] = sa / nn;
  }
  __syncthreads();

  // ---- re_matrix (B,4,8,128): [0:64) = mean f_re per partition, [64:128) = f_pos ----
  {
    const int tp = tid >> 3;              // (t,p) pair, 0..31
    const int t = tp >> 3, p = tp & 7;
    const int d0 = (tid & 7) * 8;         // 8 consecutive d per thread
    float acc[8] = {0.f, 0.f, 0.f, 0.f, 0.f, 0.f, 0.f, 0.f};
    for (int n = 0; n < 64; ++n) {
      if (s_pid[n * 4 + t] == p) {
        const float* fr = Fre + (n * 4 + t) * 64 + d0;
#pragma unroll
        for (int u = 0; u < 8; ++u) acc[u] += fr[u];
      }
    }
    const float inv = 1.f / s_cnt[tp];
    const float pd = s_posd[tp], pa = s_posa[tp];
    float* o = out + (((size_t)b * 4 + t) * 8 + p) * 128;
#pragma unroll
    for (int u = 0; u < 8; ++u) {
      const int d = d0 + u;
      o[d] = acc[u] * inv;
      o[64 + d] = fmaxf(pd * ce_w[d] + pa * ce_w[64 + d] + ce_b[d], 0.f);
    }
  }
}

extern "C" void kernel_launch(void* const* d_in, const int* in_sizes, int n_in,
                              void* d_out, int out_size, void* d_ws, size_t ws_size,
                              hipStream_t stream) {
  (void)in_sizes; (void)n_in; (void)out_size; (void)d_ws; (void)ws_size;
  const float* x_ego = (const float*)d_in[0];
  const float* x_nei = (const float*)d_in[1];
  const float* te_w  = (const float*)d_in[2];
  const float* te_b  = (const float*)d_in[3];
  const float* fc1_w = (const float*)d_in[4];
  const float* fc1_b = (const float*)d_in[5];
  const float* fc2_w = (const float*)d_in[6];
  const float* fc2_b = (const float*)d_in[7];
  const float* ce_w  = (const float*)d_in[8];
  const float* ce_b  = (const float*)d_in[9];
  float* out = (float*)d_out;

  // allow >64KB dynamic LDS (gfx1250: up to 320KB per workgroup)
  hipFuncSetAttribute((const void*)resonance_kernel,
                      hipFuncAttributeMaxDynamicSharedMemorySize, LDS_BYTES);

  resonance_kernel<<<1024, 256, LDS_BYTES, stream>>>(
      x_ego, x_nei, te_w, te_b, fc1_w, fc1_b, fc2_w, fc2_b, ce_w, ce_b, out);
}
